// LSTM_42760694399323
// MI455X (gfx1250) — matrix operand order repaired and checked by reading, not compiled, not run on hardware
//
#include <hip/hip_runtime.h>
#include <hip/hip_bf16.h>

// ---------------- problem constants ----------------
#define T_SEQ 8192
#define H     512
#define G4H   2048      // 4*H gate rows
#define HLN   20        // MLP hidden
#define NWG   64        // persistent workgroups for the recurrence
#define ROWS  32        // gate rows per WG = (H/NWG)*4 = 8 units * 4 gates

typedef __attribute__((ext_vector_type(16))) __bf16 v16bf;
typedef __attribute__((ext_vector_type(8)))  float  v8f;

// float -> bf16 with round-to-nearest-even, done in integer domain
__device__ __forceinline__ unsigned short f2bf(float f) {
  unsigned u = __builtin_bit_cast(unsigned, f);
  unsigned r = u + 0x7fffu + ((u >> 16) & 1u);
  return (unsigned short)(r >> 16);
}
// unpack packed bf16 pair (low = even k, high = odd k)
__device__ __forceinline__ float bf_lo(unsigned w) { return __builtin_bit_cast(float, w << 16); }
__device__ __forceinline__ float bf_hi(unsigned w) { return __builtin_bit_cast(float, w & 0xffff0000u); }

__device__ __forceinline__ float sigf(float x) { return 1.0f / (1.0f + __expf(-x)); }

// ---------------- init: zero h double-buffers + barrier counters ----------------
__global__ void init_kernel(float* hbufA, float* hbufB, unsigned* ctrs) {
  int i = threadIdx.x;
  for (int k = i; k < 1024; k += 256) { hbufA[k] = 0.f; hbufB[k] = 0.f; }
  if (i < 4) ctrs[i] = 0u;
}

// ---------------- pack w_ih1 into WMMA-native bf16 B-tiles ----------------
// Layout: Bpack[ntile(128)][ktile(16)][lane(32)][16 bf16]
// lane<16 holds K = ktile*32 + 0..15 of column n0+lane; lane>=16 holds K = +16..31.
__global__ __launch_bounds__(256) void bpack_kernel(const float* __restrict__ w_ih1,
                                                    unsigned short* __restrict__ Bpack) {
  int o    = blockIdx.x * 256 + threadIdx.x;     // 128*16*32*16 = 1,048,576
  int i    = o & 15;
  int lane = (o >> 4) & 31;
  int q    = (o >> 9) & 15;
  int j    = o >> 13;
  int n = j * 16 + (lane & 15);
  int k = q * 32 + ((lane >> 4) << 4) + i;
  Bpack[o] = f2bf(w_ih1[(size_t)n * H + k]);     // B[k][n] = w_ih1[n][k]
}

// ---------------- WMMA GEMM: xp1[T][2048] = h1(bf16) @ w_ih1^T + (b_ih1+b_hh1) ----------------
// grid (T/16, 2048/128), 8 waves/WG, one 16x16 C tile per wave, K looped 512 in steps of 32.
__global__ __launch_bounds__(256) void gemm_xp1_kernel(const unsigned short* __restrict__ Abf,
                                                       const unsigned short* __restrict__ Bpack,
                                                       const float* __restrict__ b_ih1,
                                                       const float* __restrict__ b_hh1,
                                                       float* __restrict__ xp1) {
  const int lane  = threadIdx.x & 31;
  const int wave  = threadIdx.x >> 5;
  const int m0    = blockIdx.x * 16;
  const int ntile = blockIdx.y * 8 + wave;
  const int n0    = ntile * 16;
  const int col   = lane & 15;
  const int hi    = lane >> 4;            // half-wave selects K half: 0..15 or 16..31 (contiguous order)

  const float bias = b_ih1[n0 + col] + b_hh1[n0 + col];
  v8f acc;
#pragma unroll
  for (int e = 0; e < 8; ++e) acc[e] = bias;   // C column n0+col gets a single bias value

  const unsigned short* arow = Abf + (size_t)(m0 + col) * H;
  const unsigned short* bptr = Bpack + (size_t)ntile * (16 * 32 * 16) + lane * 16;

#pragma unroll 4
  for (int kt = 0; kt < 16; ++kt) {
    union { v16bf v; uint4 u[2]; } a, b;
    const int kb = kt * 32 + hi * 16;           // A: lanes 0-15 K=0..15, lanes 16-31 K=16..31
    a.u[0] = *(const uint4*)(arow + kb);        // K kb..kb+7
    a.u[1] = *(const uint4*)(arow + kb + 8);    // K kb+8..kb+15
    b.u[0] = *(const uint4*)(bptr + kt * (32 * 16));
    b.u[1] = *(const uint4*)(bptr + kt * (32 * 16) + 8);
    if (kt + 1 < 16) __builtin_prefetch(bptr + (kt + 1) * (32 * 16), 0, 1);
    acc = __builtin_amdgcn_wmma_f32_16x16x32_bf16(false, a.v, false, b.v,
                                                  (short)0, acc, false, false);
  }
  // C layout: VGPR e -> row m0 + e + 8*hi, column n0+col
  float* out = xp1 + (size_t)(m0 + hi * 8) * G4H + n0 + col;
#pragma unroll
  for (int e = 0; e < 8; ++e) out[(size_t)e * G4H] = acc[e];
}

// ---------------- persistent LSTM recurrence ----------------
// 64 WGs x 256 threads; each WG owns 8 hidden units (32 gate rows).
// Weights bf16-packed into LDS once; h double-buffered in L2; one atomic grid
// barrier per step (monotonic ticket counter -> no reset races).
__global__ __launch_bounds__(256) void lstm_rec_kernel(
    const float* __restrict__ whh, const float* __restrict__ xseq,
    const float* __restrict__ wih0, const float* __restrict__ bih,
    const float* __restrict__ bhh, const float* __restrict__ xp1,
    float* __restrict__ hbuf, unsigned short* __restrict__ h1bf,
    unsigned* __restrict__ ctr, int layer) {
  __shared__ unsigned lds_w[ROWS][257];   // 257-uint row stride -> bank-staggered
  __shared__ float    lds_h[H];
  __shared__ float    lds_g[ROWS];

  const int tid = threadIdx.x;
  const int j0  = blockIdx.x * 8;

  // preload this WG's w_hh slice as packed bf16 pairs (unit-major rows: r = u*4+q)
  for (int idx = tid; idx < ROWS * 256; idx += 256) {
    int r = idx >> 8, kk = idx & 255;
    int grow = (r & 3) * H + (j0 + (r >> 2));
    float w0 = whh[(size_t)grow * H + 2 * kk];
    float w1 = whh[(size_t)grow * H + 2 * kk + 1];
    lds_w[r][kk] = (unsigned)f2bf(w0) | ((unsigned)f2bf(w1) << 16);
  }

  float c_state = 0.f;
  float xg[4] = {0.f, 0.f, 0.f, 0.f}, xb[4] = {0.f, 0.f, 0.f, 0.f};
  if (tid < 8) {   // update thread for hidden unit j0+tid
    int j = j0 + tid;
#pragma unroll
    for (int q = 0; q < 4; ++q) {
      int grow = q * H + j;
      xb[q] = bih[grow] + bhh[grow];          // layer 0 only (layer 1 bias is in xp1)
      xg[q] = (layer == 0) ? wih0[grow] : 0.f;
    }
  }
  __syncthreads();

  const int r = tid >> 3;   // gate row 0..31
  const int s = tid & 7;    // k-segment 0..7 (64 floats each)

  for (int t = 0; t < T_SEQ; ++t) {
    const int rb = t & 1, wb = rb ^ 1;
    for (int idx = tid; idx < H; idx += 256) lds_h[idx] = hbuf[rb * H + idx];
    __syncthreads();

    float acc = 0.f;
#pragma unroll 4
    for (int i = 0; i < 16; ++i) {
      int cc = (i + 2 * s) & 15;   // rotate so the 8 segments hit disjoint LDS banks
      uint2  w  = *(const uint2*)&lds_w[r][s * 32 + cc * 2];
      float4 h4 = *(const float4*)&lds_h[s * 64 + cc * 4];
      acc += bf_lo(w.x) * h4.x + bf_hi(w.x) * h4.y;
      acc += bf_lo(w.y) * h4.z + bf_hi(w.y) * h4.w;
    }
    acc += __shfl_xor(acc, 1, 32);
    acc += __shfl_xor(acc, 2, 32);
    acc += __shfl_xor(acc, 4, 32);
    if (s == 0) lds_g[r] = acc;
    __syncthreads();

    if (tid < 8) {
      int j = j0 + tid;
      float g0, g1, g2, g3;
      if (layer == 0) {
        float xt = xseq[t];
        g0 = lds_g[tid * 4 + 0] + xt * xg[0] + xb[0];
        g1 = lds_g[tid * 4 + 1] + xt * xg[1] + xb[1];
        g2 = lds_g[tid * 4 + 2] + xt * xg[2] + xb[2];
        g3 = lds_g[tid * 4 + 3] + xt * xg[3] + xb[3];
      } else {
        const float* xr = xp1 + (size_t)t * G4H;
        g0 = lds_g[tid * 4 + 0] + xr[0 * H + j];
        g1 = lds_g[tid * 4 + 1] + xr[1 * H + j];
        g2 = lds_g[tid * 4 + 2] + xr[2 * H + j];
        g3 = lds_g[tid * 4 + 3] + xr[3 * H + j];
      }
      float ig = sigf(g0), fg = sigf(g1), gg = tanhf(g2), og = sigf(g3);
      c_state = fg * c_state + ig * gg;
      float h = og * tanhf(c_state);
      hbuf[wb * H + j] = h;
      if (layer == 0) h1bf[(size_t)t * H + j] = f2bf(h);
    }

    // grid-wide step barrier (release h stores, monotonic ticket, acquire)
    __threadfence();
    __syncthreads();
    if (tid == 0) {
      __hip_atomic_fetch_add(ctr, 1u, __ATOMIC_RELEASE, __HIP_MEMORY_SCOPE_AGENT);
      const unsigned target = (unsigned)(t + 1) * (unsigned)NWG;
      while (__hip_atomic_load(ctr, __ATOMIC_ACQUIRE, __HIP_MEMORY_SCOPE_AGENT) < target)
        __builtin_amdgcn_s_sleep(1);
    }
    __syncthreads();
    __threadfence();
  }
}

// ---------------- MLP head on final h2 only ----------------
__global__ void head_kernel(const float* __restrict__ h2, const float* __restrict__ w1,
                            const float* __restrict__ b1, const float* __restrict__ w2,
                            const float* __restrict__ b2, float* __restrict__ out) {
  int lane = threadIdx.x;  // 32 threads
  float part[HLN];
#pragma unroll
  for (int o = 0; o < HLN; ++o) part[o] = 0.f;
  for (int k = lane; k < H; k += 32) {
    float hv = h2[k];
#pragma unroll
    for (int o = 0; o < HLN; ++o) part[o] += hv * w1[o * H + k];
  }
#pragma unroll
  for (int o = 0; o < HLN; ++o) {
    part[o] += __shfl_xor(part[o], 1, 32);
    part[o] += __shfl_xor(part[o], 2, 32);
    part[o] += __shfl_xor(part[o], 4, 32);
    part[o] += __shfl_xor(part[o], 8, 32);
    part[o] += __shfl_xor(part[o], 16, 32);
  }
  if (lane == 0) {
    float s = b2[0];
    for (int o = 0; o < HLN; ++o) s += (part[o] + b1[o]) * w2[o];
    out[0] = s;
  }
}

extern "C" void kernel_launch(void* const* d_in, const int* in_sizes, int n_in,
                              void* d_out, int out_size, void* d_ws, size_t ws_size,
                              hipStream_t stream) {
  const float* x    = (const float*)d_in[0];
  const float* wih0 = (const float*)d_in[1];
  const float* whh0 = (const float*)d_in[2];
  const float* bih0 = (const float*)d_in[3];
  const float* bhh0 = (const float*)d_in[4];
  const float* wih1 = (const float*)d_in[5];
  const float* whh1 = (const float*)d_in[6];
  const float* bih1 = (const float*)d_in[7];
  const float* bhh1 = (const float*)d_in[8];
  const float* w1   = (const float*)d_in[9];
  const float* b1   = (const float*)d_in[10];
  const float* w2   = (const float*)d_in[11];
  const float* b2   = (const float*)d_in[12];

  char* ws = (char*)d_ws;
  float*          xp1   = (float*)ws;                              // 64 MB fp32 [T][2048]
  unsigned short* h1bf  = (unsigned short*)(ws + 67108864);        //  8 MB bf16 [T][512]
  unsigned short* Bpack = (unsigned short*)(ws + 75497472);        //  2 MB bf16 packed B
  float*          hbufA = (float*)(ws + 77594624);                 // [2][512]
  float*          hbufB = (float*)(ws + 77598720);                 // [2][512]
  unsigned*       ctrs  = (unsigned*)(ws + 77602816);              // barrier counters

  init_kernel<<<1, 256, 0, stream>>>(hbufA, hbufB, ctrs);
  bpack_kernel<<<4096, 256, 0, stream>>>(wih1, Bpack);
  lstm_rec_kernel<<<NWG, 256, 0, stream>>>(whh0, x, wih0, bih0, bhh0, nullptr,
                                           hbufA, h1bf, ctrs + 0, 0);
  gemm_xp1_kernel<<<dim3(T_SEQ / 16, G4H / 128), 256, 0, stream>>>(h1bf, Bpack,
                                                                   bih1, bhh1, xp1);
  lstm_rec_kernel<<<NWG, 256, 0, stream>>>(whh1, nullptr, nullptr, bih1, bhh1, xp1,
                                           hbufB, nullptr, ctrs + 1, 1);
  // T even -> final step t=8191 wrote wb=0 -> final h2 is hbufB[0..511]
  head_kernel<<<1, 32, 0, stream>>>(hbufB, w1, b1, w2, b2, (float*)d_out);
}
